// DeformableBlock_6339371729455
// MI455X (gfx1250) — compile-verified, matrix operand order checked by reading, not run on hardware
//
#include <hip/hip_runtime.h>

// ---------------------------------------------------------------------------
// MI455X / gfx1250 deformable-attention block.
// All GEMM work (sample projection C->16, MLP 128->256->128) uses
// v_wmma_f32_16x16x32_f16. A-fragments are staged in LDS in the exact CDNA5
// WMMA layout (one 32B ds_load per K-chunk per lane); B-fragments (weights)
// are pre-swizzled ONCE into f16 fragment layout in workspace, so the GEMM
// hot loops do a single coalesced 32B global load per chunk instead of 16
// strided f32 loads + converts.
// ---------------------------------------------------------------------------

typedef __attribute__((ext_vector_type(16))) _Float16 v16h;
typedef __attribute__((ext_vector_type(8)))  float    v8f;

#define B_   512
#define L_   4
#define P_   17
#define D_   128
#define HID_ 256
#define T_   (B_*L_*P_)          // 34816 tokens

// CDNA5 WMMA 16x16x32 f16 A/B fragment K-mapping:
//   lane L (<16): M/N = L, element e<8 -> K = e,      e>=8 -> K = 16+(e-8)
//   lane 16+L   : M/N = L, element e<8 -> K = 8+e,    e>=8 -> K = 24+(e-8)
// Inverse (writer side): kin = K&31:
//   lane = base + ((kin>>3)&1)*16 ; e = (kin&7) + ((kin>>4)&1)*8
__device__ __forceinline__ int frag_elem_k(int kh, int e) {
  return (e < 8) ? (kh * 8 + e) : (16 + kh * 8 + (e - 8));
}

// Intra-wave LDS producer->consumer fence: lanes of the same wave wrote LDS
// that other lanes of the same wave will now read. DS ops complete in order
// per wave; drain the counter and pin the scheduler.
__device__ __forceinline__ void lds_fence_wave() {
  asm volatile("s_wait_dscnt 0x0" ::: "memory");
  __builtin_amdgcn_wave_barrier();
}

__device__ __forceinline__ float wave_sum32(float v) {
  v += __shfl_xor(v, 16);
  v += __shfl_xor(v, 8);
  v += __shfl_xor(v, 4);
  v += __shfl_xor(v, 2);
  v += __shfl_xor(v, 1);
  return v;
}

// ---------------------------------------------------------------------------
// Kernel W: pack an f32 (K x N) weight matrix into f16 WMMA-B fragment layout:
//   dst[(ntile*(K/32) + chunk)*512 + lane*16 + e] = W[k][ntile*16 + (lane&15)]
// with k = chunk*32 + frag_elem_k(lane>>4, e). One thread per output half.
// ---------------------------------------------------------------------------
__global__ void pack_bfrag_kernel(const float* __restrict__ src,
                                  _Float16* __restrict__ dst, int K, int N) {
  int idx = blockIdx.x * 256 + threadIdx.x;
  int nchunk = K >> 5;
  int total = nchunk * (N >> 4) * 512;
  if (idx >= total) return;
  int frag = idx >> 9;
  int r    = idx & 511;
  int lane = r >> 4, e = r & 15;
  int ntile = frag / nchunk, chunk = frag % nchunk;
  int n = lane & 15, kh = lane >> 4;
  int k = chunk * 32 + frag_elem_k(kh, e);
  dst[idx] = (_Float16)src[k * N + ntile * 16 + n];
}

// ---------------------------------------------------------------------------
// Kernel 0: pass-through of x[:, 0] into out[:, 0]
// ---------------------------------------------------------------------------
__global__ void copy_x0_kernel(const float* __restrict__ x, float* __restrict__ out, int n) {
  int idx = blockIdx.x * 256 + threadIdx.x;
  if (idx < n) {
    int b = idx / (P_ * D_);
    int r = idx % (P_ * D_);
    size_t off = (size_t)b * 5 * P_ * D_ + r;
    out[off] = x[off];
  }
}

// ---------------------------------------------------------------------------
// Kernel 1: per-token layernorm + attention-weight softmax + tanh offsets.
// One 128-thread block per token. Writes w_ws[T][32], pos_ws[T][32][2].
// ---------------------------------------------------------------------------
__global__ void prep_kernel(const float* __restrict__ x,
                            const float* __restrict__ ref,
                            const float* __restrict__ n1g, const float* __restrict__ n1b,
                            const float* __restrict__ aww, const float* __restrict__ awb,
                            const float* __restrict__ sow, const float* __restrict__ sob,
                            float* __restrict__ w_ws, float* __restrict__ pos_ws) {
  int t = blockIdx.x;
  int b = t / (L_ * P_);
  int l = (t / P_) % L_;
  int p = t % P_;
  int d = threadIdx.x;

  __shared__ float red[128];
  __shared__ float hl[128];
  __shared__ float awl[32];

  float xv = x[((size_t)(b * 5 + 1 + l) * P_ + p) * D_ + d] +
             x[((size_t)(b * 5) * P_ + p) * D_ + d];

  red[d] = xv; __syncthreads();
  for (int s = 64; s > 0; s >>= 1) { if (d < s) red[d] += red[d + s]; __syncthreads(); }
  float mean = red[0] * (1.0f / 128.0f);
  __syncthreads();
  float c = xv - mean;
  red[d] = c * c; __syncthreads();
  for (int s = 64; s > 0; s >>= 1) { if (d < s) red[d] += red[d + s]; __syncthreads(); }
  float var = red[0] * (1.0f / 128.0f);

  float h = c * rsqrtf(var + 1e-5f) * n1g[d] + n1b[d];
  hl[d] = h;
  __syncthreads();

  if (d < 32) {                       // attention-weight logits (128 -> 32)
    float s = awb[d];
    for (int k = 0; k < 128; ++k) s += hl[k] * aww[k * 32 + d];
    awl[d] = s;
  } else if (d < 96) {                // sampling offsets (128 -> 64), tanh, + ref
    int j = d - 32;
    float s = sob[j];
    for (int k = 0; k < 128; ++k) s += hl[k] * sow[k * 64 + j];
    float off = tanhf(s);
    pos_ws[(size_t)t * 64 + j] = off + ref[((size_t)b * P_ + p) * 2 + (j & 1)];
  }
  __syncthreads();

  if (d < 8) {                        // softmax over 4 samples per head
    float v0 = awl[d * 4], v1 = awl[d * 4 + 1], v2 = awl[d * 4 + 2], v3 = awl[d * 4 + 3];
    float m = fmaxf(fmaxf(v0, v1), fmaxf(v2, v3));
    float e0 = expf(v0 - m), e1 = expf(v1 - m), e2 = expf(v2 - m), e3 = expf(v3 - m);
    float inv = 1.0f / (e0 + e1 + e2 + e3);
    float* wo = w_ws + (size_t)t * 32 + d * 4;
    wo[0] = e0 * inv; wo[1] = e1 * inv; wo[2] = e2 * inv; wo[3] = e3 * inv;
  }
}

// ---------------------------------------------------------------------------
// Kernel 2: bilinear grid-sample + WMMA projection (C -> 16) + head-weighted
// sum + residual. One 256-thread block per (b, l); 8 waves, each wave owns a
// 16-point M-tile (34 tiles of the 17*32 = 544 sample points).
// Dynamic LDS (80512 B): a_lds 65536 | attn 8704 | wl 2176 | ptw 2048 | pto 2048
// ---------------------------------------------------------------------------
__global__ void sample_attn_kernel(const float* __restrict__ x,
                                   const float* __restrict__ w_ws,
                                   const float* __restrict__ pos_ws,
                                   const float* __restrict__ f0, const float* __restrict__ f1,
                                   const float* __restrict__ f2, const float* __restrict__ f3,
                                   const _Float16* __restrict__ e0f, const _Float16* __restrict__ e1f,
                                   const _Float16* __restrict__ e2f, const _Float16* __restrict__ e3f,
                                   const float* __restrict__ e0b, const float* __restrict__ e1b,
                                   const float* __restrict__ e2b, const float* __restrict__ e3b,
                                   float* __restrict__ out) {
  extern __shared__ unsigned char smem[];
  _Float16* a_lds = (_Float16*)smem;                   // 8*4096 halfs
  float* attn = (float*)(smem + 8 * 4096 * 2);         // 2176 f32
  float* wl   = attn + P_ * D_;                        // 544 f32
  float* ptw  = wl + P_ * 32;                          // 512 f32
  int*   pto  = (int*)(ptw + 512);                     // 512 i32

  int bl = blockIdx.x;
  int b = bl >> 2, l = bl & 3;
  int tid = threadIdx.x, lane = tid & 31, wave = tid >> 5;

  const float* F; const _Float16* EWF; const float* EB; int C, Hh, Ww;
  if      (l == 0) { F = f0; EWF = e0f; EB = e0b; C = 32;  Hh = 64; Ww = 48; }
  else if (l == 1) { F = f1; EWF = e1f; EB = e1b; C = 64;  Hh = 32; Ww = 24; }
  else if (l == 2) { F = f2; EWF = e2f; EB = e2b; C = 128; Hh = 16; Ww = 12; }
  else             { F = f3; EWF = e3f; EB = e3b; C = 256; Hh = 8;  Ww = 6;  }
  const int HW = Hh * Ww;
  const float* Fb = F + (size_t)b * C * HW;            // per-batch slice: L2-resident
  size_t tokbase = (size_t)bl * P_;

  if (lane == 0) __builtin_prefetch(Fb + wave * (C * HW / 8), 0, 1);  // warm L2

  for (int i = tid; i < P_ * 32; i += 256) wl[i] = w_ws[tokbase * 32 + i];
  __syncthreads();

  const int nchunk = C >> 5;
  _Float16* aw    = a_lds + wave * 4096;
  float*    myptw = ptw + wave * 64;
  int*      mypto = pto + wave * 64;
  const int n  = lane & 15;
  const int kh = lane >> 4;

  for (int tile = wave; tile < 34; tile += 8) {
    // lanes 0..15: bilinear params for the tile's 16 points
    if (lane < 16) {
      int gpt = tile * 16 + lane;
      float px = pos_ws[(tokbase * 32 + gpt) * 2 + 0];
      float py = pos_ws[(tokbase * 32 + gpt) * 2 + 1];
      float gx = (px + 1.0f) * 0.5f * (float)(Ww - 1);
      float gy = (py + 1.0f) * 0.5f * (float)(Hh - 1);
      float x0f = floorf(gx), y0f = floorf(gy);
      float wx = gx - x0f, wy = gy - y0f;
      int x0i = (int)fminf(fmaxf(x0f,        0.f), (float)(Ww - 1));
      int x1i = (int)fminf(fmaxf(x0f + 1.f,  0.f), (float)(Ww - 1));
      int y0i = (int)fminf(fmaxf(y0f,        0.f), (float)(Hh - 1));
      int y1i = (int)fminf(fmaxf(y0f + 1.f,  0.f), (float)(Hh - 1));
      myptw[lane * 4 + 0] = (1.f - wx) * (1.f - wy);
      myptw[lane * 4 + 1] = wx * (1.f - wy);
      myptw[lane * 4 + 2] = (1.f - wx) * wy;
      myptw[lane * 4 + 3] = wx * wy;
      mypto[lane * 4 + 0] = y0i * Ww + x0i;
      mypto[lane * 4 + 1] = y0i * Ww + x1i;
      mypto[lane * 4 + 2] = y1i * Ww + x0i;
      mypto[lane * 4 + 3] = y1i * Ww + x1i;
    }
    lds_fence_wave();

    // Hoist this lane's point params: m = i&15 is loop-invariant per lane.
    const int m = n;                          // point row within tile
    float w00 = myptw[m * 4 + 0], w01 = myptw[m * 4 + 1];
    float w10 = myptw[m * 4 + 2], w11 = myptw[m * 4 + 3];
    int   o00 = mypto[m * 4 + 0], o01 = mypto[m * 4 + 1];
    int   o10 = mypto[m * 4 + 2], o11 = mypto[m * 4 + 3];

    // gather 16 points x C channels straight into WMMA-A layout
    // (lane m handles even channels, lane 16+m odd channels)
    for (int c = kh; c < C; c += 2) {
      const float* fc = Fb + (size_t)c * HW;
      float v = w00 * fc[o00] + w01 * fc[o01] + w10 * fc[o10] + w11 * fc[o11];
      int kin = c & 31, chunk = c >> 5;
      int lt = m + ((kin >> 3) & 1) * 16;
      int e  = (kin & 7) + ((kin >> 4) & 1) * 8;
      aw[chunk * 512 + lt * 16 + e] = (_Float16)v;
    }
    lds_fence_wave();

    // projection: (16 x C) @ (C x 16) via WMMA; B-frags pre-packed in global
    v8f acc = {};
    for (int chunk = 0; chunk < nchunk; ++chunk) {
      v16h A  = *(const v16h*)(aw + chunk * 512 + lane * 16);
      v16h Bf = *(const v16h*)(EWF + chunk * 512 + lane * 16);
      acc = __builtin_amdgcn_wmma_f32_16x16x32_f16(false, A, false, Bf,
                                                   (short)0, acc, false, false);
    }

    // fold softmax weights; each (token, head, col) is owned by exactly one
    // lane (col = n, heads split by kh and r-group), so plain stores suffice.
    int tok = tile >> 1, halfsel = tile & 1;
    int mbase = kh * 8;
    float s0 = 0.f, s1 = 0.f;
    for (int r = 0; r < 4; ++r) {
      s0 += wl[tok * 32 + halfsel * 16 + mbase + r]     * acc[r];
      s1 += wl[tok * 32 + halfsel * 16 + mbase + 4 + r] * acc[4 + r];
    }
    int h0 = halfsel * 4 + (mbase >> 2);
    float eb = EB[n];                  // softmax sums to 1 -> bias added once
    attn[tok * D_ + h0 * 16 + n]       = s0 + eb;
    attn[tok * D_ + (h0 + 1) * 16 + n] = s1 + eb;
  }
  __syncthreads();

  size_t obase = (size_t)((b * 5 + 1 + l) * P_) * D_;
  for (int i = tid; i < P_ * D_; i += 256)
    out[obase + i] = x[obase + i] + attn[i];   // y = residual + attn
}

// ---------------------------------------------------------------------------
// Kernel 3: layernorm + MLP (128 -> 256, exact GELU, 256 -> 128) + residual.
// One 256-thread block per (b, l): M = 17 tokens (2 M-tiles, padded).
// ---------------------------------------------------------------------------
__global__ void mlp_kernel(const float* __restrict__ n2g, const float* __restrict__ n2b,
                           const _Float16* __restrict__ fc1f, const float* __restrict__ fc1b,
                           const _Float16* __restrict__ fc2f, const float* __restrict__ fc2b,
                           float* __restrict__ out) {
  __shared__ float    yl[P_ * D_];          // y (pre-MLP)
  __shared__ _Float16 a1[2 * 4 * 512];      // GEMM1 A (K = 128)
  __shared__ _Float16 a2[2 * 8 * 512];      // GEMM2 A (K = 256)

  int bl = blockIdx.x;
  int b = bl >> 2, l = bl & 3;
  int tid = threadIdx.x, lane = tid & 31, wave = tid >> 5;
  size_t ybase = (size_t)((b * 5 + 1 + l) * P_) * D_;

  for (int i = tid; i < P_ * D_; i += 256) yl[i] = out[ybase + i];
  __syncthreads();

  // layernorm per token, write into WMMA-A layout (f16)
  for (int tk = wave; tk < P_; tk += 8) {
    float v0 = yl[tk * D_ + lane * 4 + 0], v1 = yl[tk * D_ + lane * 4 + 1];
    float v2 = yl[tk * D_ + lane * 4 + 2], v3 = yl[tk * D_ + lane * 4 + 3];
    float mean = wave_sum32(v0 + v1 + v2 + v3) * (1.0f / 128.0f);
    float c0 = v0 - mean, c1 = v1 - mean, c2 = v2 - mean, c3 = v3 - mean;
    float var = wave_sum32(c0 * c0 + c1 * c1 + c2 * c2 + c3 * c3) * (1.0f / 128.0f);
    float rs = rsqrtf(var + 1e-5f);
    int mrow = tk & 15, mtile = tk >> 4;
    float cv[4] = {c0, c1, c2, c3};
    for (int q = 0; q < 4; ++q) {
      int k = lane * 4 + q;
      float h = cv[q] * rs * n2g[k] + n2b[k];
      int kin = k & 31, chunk = k >> 5;
      int lt = mrow + ((kin >> 3) & 1) * 16;
      int e  = (kin & 7) + ((kin >> 4) & 1) * 8;
      a1[(mtile * 4 + chunk) * 512 + lt * 16 + e] = (_Float16)h;
    }
  }
  __syncthreads();

  const int n  = lane & 15;
  const int kh = lane >> 4;

  // GEMM1: 2 M-tiles x 16 N-tiles, K = 128; epilogue: bias + exact GELU -> a2
  for (int job = wave; job < 32; job += 8) {
    int mtile = job >> 4, ntile = job & 15;
    v8f acc = {};
    for (int chunk = 0; chunk < 4; ++chunk) {
      v16h A  = *(const v16h*)(a1 + (mtile * 4 + chunk) * 512 + lane * 16);
      v16h Bf = *(const v16h*)(fc1f + (ntile * 4 + chunk) * 512 + lane * 16);
      acc = __builtin_amdgcn_wmma_f32_16x16x32_f16(false, A, false, Bf,
                                                   (short)0, acc, false, false);
    }
    int col = ntile * 16 + n;
    float bias = fc1b[col];
    for (int r = 0; r < 8; ++r) {
      int m = mtile * 16 + kh * 8 + r;
      if (m < P_) {
        float v = acc[r] + bias;
        float g = 0.5f * v * (1.0f + erff(v * 0.70710678118f));   // exact GELU
        int kin = col & 31, chunk2 = col >> 5;
        int lt = (m & 15) + ((kin >> 3) & 1) * 16;
        int e  = (kin & 7) + ((kin >> 4) & 1) * 8;
        a2[((m >> 4) * 8 + chunk2) * 512 + lt * 16 + e] = (_Float16)g;
      }
    }
  }
  __syncthreads();

  // GEMM2: 2 M-tiles x 8 N-tiles, K = 256; epilogue: bias + residual -> out
  for (int job = wave; job < 16; job += 8) {
    int mtile = job >> 3, ntile = job & 7;
    v8f acc = {};
    for (int chunk = 0; chunk < 8; ++chunk) {
      v16h A  = *(const v16h*)(a2 + (mtile * 8 + chunk) * 512 + lane * 16);
      v16h Bf = *(const v16h*)(fc2f + (ntile * 8 + chunk) * 512 + lane * 16);
      acc = __builtin_amdgcn_wmma_f32_16x16x32_f16(false, A, false, Bf,
                                                   (short)0, acc, false, false);
    }
    int col = ntile * 16 + n;
    float bias = fc2b[col];
    for (int r = 0; r < 8; ++r) {
      int m = mtile * 16 + kh * 8 + r;
      if (m < P_) out[ybase + m * D_ + col] = yl[m * D_ + col] + acc[r] + bias;
    }
  }
}

// ---------------------------------------------------------------------------
extern "C" void kernel_launch(void* const* d_in, const int* in_sizes, int n_in,
                              void* d_out, int out_size, void* d_ws, size_t ws_size,
                              hipStream_t stream) {
  (void)in_sizes; (void)n_in; (void)out_size; (void)ws_size;
  const float* x    = (const float*)d_in[0];
  const float* ref  = (const float*)d_in[1];
  const float* f0   = (const float*)d_in[2];
  const float* f1   = (const float*)d_in[3];
  const float* f2   = (const float*)d_in[4];
  const float* f3   = (const float*)d_in[5];
  const float* n1g  = (const float*)d_in[6];
  const float* n1b  = (const float*)d_in[7];
  const float* aww  = (const float*)d_in[8];
  const float* awb  = (const float*)d_in[9];
  const float* sow  = (const float*)d_in[10];
  const float* sob  = (const float*)d_in[11];
  const float* e0w  = (const float*)d_in[12];
  const float* e0b  = (const float*)d_in[13];
  const float* e1w  = (const float*)d_in[14];
  const float* e1b  = (const float*)d_in[15];
  const float* e2w  = (const float*)d_in[16];
  const float* e2b  = (const float*)d_in[17];
  const float* e3w  = (const float*)d_in[18];
  const float* e3b  = (const float*)d_in[19];
  const float* n2g  = (const float*)d_in[20];
  const float* n2b  = (const float*)d_in[21];
  const float* fc1w = (const float*)d_in[22];
  const float* fc1b = (const float*)d_in[23];
  const float* fc2w = (const float*)d_in[24];
  const float* fc2b = (const float*)d_in[25];
  float* out = (float*)d_out;

  // workspace layout
  float* w_ws   = (float*)d_ws;                          // T*32 f32
  float* pos_ws = w_ws + (size_t)T_ * 32;                // T*64 f32
  _Float16* fc1f = (_Float16*)(pos_ws + (size_t)T_ * 64);// 32768 halfs
  _Float16* fc2f = fc1f + 32768;                         // 32768 halfs
  _Float16* e0f  = fc2f + 32768;                         // 512
  _Float16* e1f  = e0f + 512;                            // 1024
  _Float16* e2f  = e1f + 1024;                           // 2048
  _Float16* e3f  = e2f + 2048;                           // 4096

  // one-shot weight pack into WMMA-B fragment layout (f16)
  pack_bfrag_kernel<<<128, 256, 0, stream>>>(fc1w, fc1f, 128, 256);
  pack_bfrag_kernel<<<128, 256, 0, stream>>>(fc2w, fc2f, 256, 128);
  pack_bfrag_kernel<<<2,   256, 0, stream>>>(e0w,  e0f,  32,  16);
  pack_bfrag_kernel<<<4,   256, 0, stream>>>(e1w,  e1f,  64,  16);
  pack_bfrag_kernel<<<8,   256, 0, stream>>>(e2w,  e2f,  128, 16);
  pack_bfrag_kernel<<<16,  256, 0, stream>>>(e3w,  e3f,  256, 16);

  int n0 = B_ * P_ * D_;
  copy_x0_kernel<<<(n0 + 255) / 256, 256, 0, stream>>>(x, out, n0);

  prep_kernel<<<T_, 128, 0, stream>>>(x, ref, n1g, n1b, aww, awb, sow, sob,
                                      w_ws, pos_ws);

  sample_attn_kernel<<<B_ * L_, 256, 80512, stream>>>(
      x, w_ws, pos_ws, f0, f1, f2, f3,
      e0f, e1f, e2f, e3f, e0b, e1b, e2b, e3b, out);

  mlp_kernel<<<B_ * L_, 256, 0, stream>>>(n2g, n2b, fc1f, fc1b, fc2f, fc2b, out);
}
